// BatchMultiHeadGraphAttention_42949672961876
// MI455X (gfx1250) — matrix operand order repaired and checked by reading, not compiled, not run on hardware
//
#include <hip/hip_runtime.h>
#include <hip/hip_bf16.h>

typedef _Float16 v16h __attribute__((ext_vector_type(16)));
typedef float    v8f  __attribute__((ext_vector_type(8)));
typedef float    f4   __attribute__((ext_vector_type(4)));
typedef unsigned int       u32;
typedef unsigned long long u64;
typedef u32 u32x4 __attribute__((ext_vector_type(4)));

#define N_HEAD 4
#define F_IN   256
#define F_OUT  64
#define BS     4
#define NN     2048
#define NEG_SLOPE 0.2f

// ---------------------------------------------------------------------------
// Kernel A: h_prime = h @ w  (per b,head), fused tanh-dot for attn_src/attn_dst.
// grid = 4*4*16 blocks of 256 threads (8 waves); each wave owns one 16x64 tile.
// ---------------------------------------------------------------------------
__global__ __launch_bounds__(256) void gat_hprime_kernel(
    const float* __restrict__ h, const float* __restrict__ w,
    const float* __restrict__ a_src, const float* __restrict__ a_dst,
    _Float16* __restrict__ hp, float* __restrict__ asrc_out,
    float* __restrict__ adst_out)
{
  __shared__ _Float16 sh_hp[8 * 16 * F_OUT];   // 16 KB repack buffer

  const int lane = threadIdx.x & 31;
  const int wid  = threadIdx.x >> 5;
  const int bh   = blockIdx.x >> 4;            // 0..15
  const int b    = bh >> 2, hd = bh & 3;
  const int tile = (blockIdx.x & 15) * 8 + wid;  // 0..127
  const int row  = lane & 15;                  // M within half
  const int half = lane >> 4;

  const float* hrow = h + ((size_t)b * NN + (size_t)tile * 16 + row) * F_IN;
  const float* wmat = w + (size_t)hd * F_IN * F_OUT;

  v8f acc[4] = {v8f{}, v8f{}, v8f{}, v8f{}};

  for (int kc = 0; kc < F_IN; kc += 32) {
    // A fragment (16x32 f16): lanes 0-15 cover K=[0,8)+[16,24), lanes 16-31 shift by 8
    f4 a0 = *(const f4*)(hrow + kc + half * 8);
    f4 a1 = *(const f4*)(hrow + kc + half * 8 + 4);
    f4 a2 = *(const f4*)(hrow + kc + 16 + half * 8);
    f4 a3 = *(const f4*)(hrow + kc + 16 + half * 8 + 4);
    float av[16] = {a0.x, a0.y, a0.z, a0.w, a1.x, a1.y, a1.z, a1.w,
                    a2.x, a2.y, a2.z, a2.w, a3.x, a3.y, a3.z, a3.w};
    v16h af;
#pragma unroll
    for (int i = 0; i < 16; ++i) af[i] = (_Float16)av[i];

    const float* wr = wmat + (size_t)kc * F_OUT + (lane & 15);
#pragma unroll
    for (int o = 0; o < 4; ++o) {
      // B fragment (32x16 f16): lane%16 = column o*16+n, element i = K in table order (same as A)
      float bv[16];
#pragma unroll
      for (int i = 0; i < 16; ++i)
        bv[i] = wr[(size_t)((i < 8 ? 0 : 8) + half * 8 + i) * F_OUT + o * 16];
      v16h bf;
#pragma unroll
      for (int i = 0; i < 16; ++i) bf[i] = (_Float16)bv[i];
      acc[o] = __builtin_amdgcn_wmma_f32_16x16x32_f16(
          false, af, false, bf, (short)0, acc[o], false, false);
    }
  }

  // attn_src/attn_dst partials: C layout lane=(col within o-tile), VGPR r -> M = half*8+r
  const int col0 = lane & 15;
  float asv[4], adv[4];
#pragma unroll
  for (int o = 0; o < 4; ++o) {
    asv[o] = a_src[hd * F_OUT + o * 16 + col0];
    adv[o] = a_dst[hd * F_OUT + o * 16 + col0];
  }
  float sp[8], dp[8];
#pragma unroll
  for (int r = 0; r < 8; ++r) { sp[r] = 0.f; dp[r] = 0.f; }
#pragma unroll
  for (int o = 0; o < 4; ++o)
#pragma unroll
    for (int r = 0; r < 8; ++r) {
      float t = tanhf(acc[o][r]);
      sp[r] += t * asv[o];
      dp[r] += t * adv[o];
    }
  // reduce across the 16 lanes of each half (rows differ between halves)
#pragma unroll
  for (int m = 1; m <= 8; m <<= 1)
#pragma unroll
    for (int r = 0; r < 8; ++r) {
      sp[r] += __shfl_xor(sp[r], m, 32);
      dp[r] += __shfl_xor(dp[r], m, 32);
    }
  if (col0 == 0) {
    const int rbase = bh * NN + tile * 16 + half * 8;
#pragma unroll
    for (int r = 0; r < 8; ++r) {
      asrc_out[rbase + r] = sp[r];
      adst_out[rbase + r] = dp[r];
    }
  }

  // repack h_prime tile to f16 through LDS, then coalesced store
  _Float16* my = sh_hp + wid * (16 * F_OUT);
#pragma unroll
  for (int o = 0; o < 4; ++o)
#pragma unroll
    for (int r = 0; r < 8; ++r)
      my[(half * 8 + r) * F_OUT + o * 16 + col0] = (_Float16)acc[o][r];
  __syncthreads();
  const u32x4* srcv = (const u32x4*)my;
  u32x4* dstv = (u32x4*)(hp + (size_t)bh * NN * F_OUT + (size_t)tile * 16 * F_OUT);
#pragma unroll
  for (int it = 0; it < 4; ++it) dstv[it * 32 + lane] = srcv[it * 32 + lane];
}

// ---------------------------------------------------------------------------
// Kernel B: fused masked softmax + (attn @ h_prime), flash-style with exact max.
// grid = 16*(2048/16) = 2048 blocks of 256 threads; block = one 16-row tile.
// Adjacency tile + dst row are staged with GLOBAL_LOAD_ASYNC_TO_LDS_B128
// (CDNA5 async copy engine, ASYNCcnt) — no VGPR round-trip.
// ---------------------------------------------------------------------------
__global__ __launch_bounds__(256) void gat_attn_kernel(
    const unsigned char* __restrict__ adj, const _Float16* __restrict__ hp,
    const float* __restrict__ asrc, const float* __restrict__ adst,
    const float* __restrict__ bias, float* __restrict__ out)
{
  __shared__ unsigned char s_adj[16 * NN];   // 32 KB adjacency tile
  __shared__ float s_dst[NN];                // 8 KB
  __shared__ float s_acc[16 * F_OUT];        // 4 KB combine buffer
  __shared__ float s_pmax[256];
  __shared__ float s_max[16], s_l[16], s_src[16];

  const int t    = threadIdx.x;
  const int lane = t & 31, wid = t >> 5;
  const int tile = blockIdx.x & 127;
  const int bh   = blockIdx.x >> 7;
  const int b    = bh >> 2;
  const int row0 = tile * 16;

  // ---- stage 0: async-stage adjacency rows + dst row into LDS, init accum
  {
    const unsigned char* gadj = adj + (size_t)b * NN * NN + (size_t)row0 * NN;
    // 32 KB = 2048 x 16B chunks, 8 per thread
    for (int i = t; i < (16 * NN) / 16; i += 256) {
      u32 ldsoff = (u32)(size_t)(s_adj + i * 16);
      const unsigned char* ga = gadj + i * 16;
      asm volatile("global_load_async_to_lds_b128 %0, %1, off"
                   :: "v"(ldsoff), "v"(ga) : "memory");
    }
    const unsigned char* gdst = (const unsigned char*)(adst + (size_t)bh * NN);
    // 8 KB = 512 x 16B chunks, 2 per thread
    for (int i = t; i < (NN * 4) / 16; i += 256) {
      u32 ldsoff = (u32)(size_t)((unsigned char*)s_dst + i * 16);
      const unsigned char* ga = gdst + i * 16;
      asm volatile("global_load_async_to_lds_b128 %0, %1, off"
                   :: "v"(ldsoff), "v"(ga) : "memory");
    }
    if (t < 16) {
      s_src[t] = asrc[bh * NN + row0 + t];
      s_l[t] = 0.f;
    }
    for (int i = t; i < 16 * F_OUT; i += 256) s_acc[i] = 0.f;
    asm volatile("s_wait_asynccnt 0" ::: "memory");
  }
  __syncthreads();

  // ---- stage 1: exact row max = leakyrelu(src_i + max_{adj} dst_j)
  {
    const int r = t & 15, seg = t >> 4;
    float m = -__builtin_inff();
    const unsigned char* arow = s_adj + r * NN + seg * 128;
    const float* drow = s_dst + seg * 128;
    for (int j = 0; j < 128; j += 4) {
      u32 a4 = *(const u32*)(arow + j);
      f4 d4 = *(const f4*)(drow + j);
      if (a4 & 0x000000ffu) m = fmaxf(m, d4.x);
      if (a4 & 0x0000ff00u) m = fmaxf(m, d4.y);
      if (a4 & 0x00ff0000u) m = fmaxf(m, d4.z);
      if (a4 & 0xff000000u) m = fmaxf(m, d4.w);
    }
    s_pmax[t] = m;
  }
  __syncthreads();
  if (t < 16) {
    float m = -__builtin_inff();
    for (int seg = 0; seg < 16; ++seg) m = fmaxf(m, s_pmax[seg * 16 + t]);
    float s = s_src[t] + m;
    s_max[t] = (s >= 0.f) ? s : NEG_SLOPE * s;
  }
  __syncthreads();

  // ---- stage 2: split-K P x Hp with WMMA; exact max means partials just add
  const int row  = lane & 15;
  const int half = lane >> 4;
  const float srcv = s_src[row];
  const float mrow = s_max[row];
  const _Float16* hpb = hp + (size_t)bh * NN * F_OUT;

  v8f acc[4] = {v8f{}, v8f{}, v8f{}, v8f{}};
  float lsum = 0.f;

  for (int c = 0; c < 8; ++c) {
    const int jc = c * 256 + wid * 32;     // this wave's K chunk
    if (c < 7)                              // warm L2/L0 for next B fragment row
      __builtin_prefetch(hpb + (size_t)(jc + 256 + lane) * F_OUT, 0, 3);
    const int jA = jc + half * 8;          // K for elements 0..7 of A fragment
    const int jB = jc + 16 + half * 8;     // K for elements 8..15
    u64 adjA = *(const u64*)(s_adj + row * NN + jA);
    u64 adjB = *(const u64*)(s_adj + row * NN + jB);
    f4 dA0 = *(const f4*)(s_dst + jA);
    f4 dA1 = *(const f4*)(s_dst + jA + 4);
    f4 dB0 = *(const f4*)(s_dst + jB);
    f4 dB1 = *(const f4*)(s_dst + jB + 4);
    float dv[16] = {dA0.x, dA0.y, dA0.z, dA0.w, dA1.x, dA1.y, dA1.z, dA1.w,
                    dB0.x, dB0.y, dB0.z, dB0.w, dB1.x, dB1.y, dB1.z, dB1.w};
    v16h pf;
#pragma unroll
    for (int i = 0; i < 16; ++i) {
      u64 ab = (i < 8) ? (adjA >> (8 * i)) : (adjB >> (8 * (i - 8)));
      float s = srcv + dv[i];
      s = (s >= 0.f) ? s : NEG_SLOPE * s;
      float p = (ab & 0xffull) ? __expf(s - mrow) : 0.f;
      lsum += p;
      pf[i] = (_Float16)p;
    }
#pragma unroll
    for (int o = 0; o < 4; ++o) {
      // B fragment: column o*16+lane%16 of h_prime, element i = K in table order (same as A)
      v16h bf;
#pragma unroll
      for (int i = 0; i < 16; ++i)
        bf[i] = hpb[(size_t)(jc + (i < 8 ? 0 : 8) + half * 8 + i) * F_OUT + o * 16 + row];
      acc[o] = __builtin_amdgcn_wmma_f32_16x16x32_f16(
          false, pf, false, bf, (short)0, acc[o], false, false);
    }
  }

  atomicAdd(&s_l[row], lsum);
#pragma unroll
  for (int o = 0; o < 4; ++o)
#pragma unroll
    for (int r = 0; r < 8; ++r)
      atomicAdd(&s_acc[(half * 8 + r) * F_OUT + o * 16 + row], acc[o][r]);
  __syncthreads();

  // ---- stage 3: normalize, add bias, store (coalesced float4)
  {
    const int idx = t * 4;
    const int r = idx >> 6, cc = idx & 63;
    float l = s_l[r];
    float inv = (l > 0.f) ? 1.f / l : 0.f;
    f4 v = *(const f4*)(s_acc + idx);
    f4 bv = *(const f4*)(bias + cc);
    f4 o4 = {v.x * inv + bv.x, v.y * inv + bv.y,
             v.z * inv + bv.z, v.w * inv + bv.w};
    *(f4*)(out + ((size_t)bh * NN + row0) * F_OUT + idx) = o4;
  }
}

// ---------------------------------------------------------------------------
extern "C" void kernel_launch(void* const* d_in, const int* in_sizes, int n_in,
                              void* d_out, int out_size, void* d_ws, size_t ws_size,
                              hipStream_t stream) {
  const float*         h     = (const float*)d_in[0];
  const unsigned char* adj   = (const unsigned char*)d_in[1];  // jax bool_
  const float*         w     = (const float*)d_in[2];
  const float*         a_src = (const float*)d_in[3];
  const float*         a_dst = (const float*)d_in[4];
  const float*         bias  = (const float*)d_in[5];
  float*               out   = (float*)d_out;

  const size_t hp_bytes = (size_t)16 * NN * F_OUT * sizeof(_Float16);  // 4 MB
  _Float16* hp   = (_Float16*)d_ws;
  float*    asrc = (float*)((char*)d_ws + hp_bytes);
  float*    adst = (float*)((char*)d_ws + hp_bytes + (size_t)16 * NN * sizeof(float));

  gat_hprime_kernel<<<dim3(256), dim3(256), 0, stream>>>(h, w, a_src, a_dst,
                                                         hp, asrc, adst);
  gat_attn_kernel<<<dim3(2048), dim3(256), 0, stream>>>(adj, hp, asrc, adst,
                                                        bias, out);
}
